// Model_33285996544307
// MI455X (gfx1250) — matrix operand order repaired and checked by reading, not compiled, not run on hardware
//
#include <hip/hip_runtime.h>
#include <hip/hip_bf16.h>

typedef __attribute__((ext_vector_type(16))) _Float16 v16h;
typedef __attribute__((ext_vector_type(8)))  float    v8f;
typedef __attribute__((ext_vector_type(2)))  float    v2f;

#define B_BATCH 4096
#define T_STEPS 128
#define E_DIM   300
#define D_DIM   25
#define C_DIM   5

#define KCH     10      // K chunks of 32 (300 padded to 320)
#define NTILES  40      // N tiles of 16 (625 padded to 640)
#define NREAL   (D_DIM * D_DIM)   // 625
#define GSTRIDE 644     // f32 row stride: 8*644 % 64 == 32 -> conflict-free half-wave offset
#define MT      32      // batch rows per block (2 m-subtiles of 16)
#define TPB     320     // 10 waves of 32

// ---------------------------------------------------------------------------
// Repack A [25,300,25] f32 -> f16 B-fragments for v_wmma_f32_16x16x32_f16.
// Fragment order: [kchunk c][ntile nt][lane][slot];  B (32x16 KxN) layout:
//   n_local = lane & 15,  k_local = slot + 16*(lane>>4)   (contiguous k order, same as ldsX)
// ---------------------------------------------------------------------------
__global__ void repack_A(const float* __restrict__ A, _Float16* __restrict__ Abf) {
  int idx = blockIdx.x * blockDim.x + threadIdx.x;
  if (idx >= KCH * NTILES * 32 * 16) return;
  int slot  = idx & 15;
  int rest  = idx >> 4;
  int lane  = rest & 31;
  int rest2 = rest >> 5;
  int nt    = rest2 % NTILES;
  int c     = rest2 / NTILES;

  int n_local = lane & 15;
  int k_local = slot + ((lane >> 4) << 4);
  int e = c * 32 + k_local;        // K dim = embedding index
  int n = nt * 16 + n_local;       // N dim = k*25 + j
  float v = 0.f;
  if (e < E_DIM && n < NREAL) {
    int kk = n / D_DIM;
    int j  = n % D_DIM;
    v = A[(kk * E_DIM + e) * D_DIM + j];
  }
  Abf[idx] = (_Float16)v;
}

// ---------------------------------------------------------------------------
// Fused RNN: per block = 32 batch rows for the full T loop.
// ---------------------------------------------------------------------------
__global__ __launch_bounds__(TPB) void rnn_fused(
    const int*   __restrict__ words,   // [B, T]
    const float* __restrict__ emb,     // [V, E]
    const _Float16* __restrict__ Abf,  // packed B fragments (L2-resident, 400 KB)
    const float* __restrict__ Wout,    // [C, D]
    const float* __restrict__ bout,    // [C]
    float*       __restrict__ out) {   // [B, C]

  __shared__ __align__(32) _Float16 ldsX[KCH][2][32][16]; // 20 KB, A-frag layout x 2 m-subtiles
  __shared__ float Gt[MT][GSTRIDE];                       // ~82 KB
  __shared__ float hcur[MT][32];                          // 4 KB

  const int tid  = threadIdx.x;
  const int lane = tid & 31;
  const int w    = tid >> 5;          // wave 0..9, owns N strip [w*64, w*64+64)
  const int b0   = blockIdx.x * MT;

  for (int i = tid; i < MT * 32; i += TPB) ((float*)hcur)[i] = 0.f;
  __syncthreads();

  // staging geometry: 10 threads per batch row, 32 consecutive e per thread
  const int sm  = tid / 10;           // row 0..31
  const int se0 = (tid % 10) * 32;    // e base 0,32,...,288
  const int msub = sm >> 4, mloc = sm & 15;

  // opaque zero offset: defeats LICM/CSE of B-fragment loads without
  // destroying the global address-space of Abf (keeps global_load_b128)
  int bofs = 0;

  for (int t = 0; t < T_STEPS; ++t) {
    // ---- stage x_t tile into LDS, pre-swizzled to WMMA A-fragment layout ----
    {
      int word = words[(b0 + sm) * T_STEPS + t];
      const float* row = emb + (long)word * E_DIM;
      if (t + 1 < T_STEPS) {   // gfx1250 global_prefetch_b8 for next step's row
        int nw = words[(b0 + sm) * T_STEPS + t + 1];
        __builtin_prefetch(emb + (long)nw * E_DIM + se0, 0, 1);
      }
#pragma unroll
      for (int i = 0; i < 16; ++i) {
        int e = se0 + 2 * i;               // even; pair (e, e+1) shares lane, slots s,s+1
        float f0 = 0.f, f1 = 0.f;
        if (e < E_DIM) {                   // E_DIM even -> pair fully in/out of range
          v2f f = *(const v2f*)&row[e];
          f0 = f.x; f1 = f.y;
        }
        int c  = e >> 5;
        int k  = e & 31;
        int l2 = mloc + ((k >> 4) << 4);              // A-frag lane (contiguous k = 16h + s)
        int s2 = k & 15;                              // A-frag slot (even), contiguous k order
        auto p = __builtin_amdgcn_cvt_pkrtz(f0, f1);  // v_cvt_pk_f16_f32, __fp16 x2
        *(decltype(p)*)&ldsX[c][msub][l2][s2] = p;    // one ds_store_b32
      }
    }
    __syncthreads();

    asm volatile("" : "+s"(bofs));   // bofs == 0, but compiler must reload B each t
    const v16h* __restrict__ Bf = ((const v16h*)Abf) + bofs;

    // ---- G strips = x_t(32x320) @ A(320x64-strip) via WMMA ----
    v8f a00 = {}, a01 = {}, a02 = {}, a03 = {};
    v8f a10 = {}, a11 = {}, a12 = {}, a13 = {};
#pragma unroll 2
    for (int c = 0; c < KCH; ++c) {
      v16h am0 = *(const v16h*)&ldsX[c][0][lane][0];
      v16h am1 = *(const v16h*)&ldsX[c][1][lane][0];
      int base = (c * NTILES + w * 4) * 32 + lane;
      v16h b0v = Bf[base];
      v16h b1v = Bf[base + 32];
      v16h b2v = Bf[base + 64];
      v16h b3v = Bf[base + 96];
      a00 = __builtin_amdgcn_wmma_f32_16x16x32_f16(false, am0, false, b0v, (short)0, a00, false, false);
      a10 = __builtin_amdgcn_wmma_f32_16x16x32_f16(false, am1, false, b0v, (short)0, a10, false, false);
      a01 = __builtin_amdgcn_wmma_f32_16x16x32_f16(false, am0, false, b1v, (short)0, a01, false, false);
      a11 = __builtin_amdgcn_wmma_f32_16x16x32_f16(false, am1, false, b1v, (short)0, a11, false, false);
      a02 = __builtin_amdgcn_wmma_f32_16x16x32_f16(false, am0, false, b2v, (short)0, a02, false, false);
      a12 = __builtin_amdgcn_wmma_f32_16x16x32_f16(false, am1, false, b2v, (short)0, a12, false, false);
      a03 = __builtin_amdgcn_wmma_f32_16x16x32_f16(false, am0, false, b3v, (short)0, a03, false, false);
      a13 = __builtin_amdgcn_wmma_f32_16x16x32_f16(false, am1, false, b3v, (short)0, a13, false, false);
    }
    // D layout (ISA 7.12.2): vgpr r -> m = r + 8*(lane>=16), n = lane&15
    {
      int nb = w * 64 + (lane & 15);
      int mb = (lane >> 4) << 3;
#pragma unroll
      for (int r = 0; r < 8; ++r) {
        Gt[mb + r][nb]           = a00[r];
        Gt[mb + r][nb + 16]      = a01[r];
        Gt[mb + r][nb + 32]      = a02[r];
        Gt[mb + r][nb + 48]      = a03[r];
        Gt[16 + mb + r][nb]      = a10[r];
        Gt[16 + mb + r][nb + 16] = a11[r];
        Gt[16 + mb + r][nb + 32] = a12[r];
        Gt[16 + mb + r][nb + 48] = a13[r];
      }
    }
    __syncthreads();

    // ---- batched matvec: h_new[m][k] = tanh( sum_j G[m][k*25+j] * h[m][j] ) ----
    float r0 = 0.f, r1 = 0.f, r2 = 0.f;
    {
      int m = tid / D_DIM, k = tid % D_DIM;       // task 0: tid < 320 < 800
      float s = 0.f;
#pragma unroll
      for (int j = 0; j < D_DIM; ++j) s += Gt[m][k * D_DIM + j] * hcur[m][j];
      r0 = tanhf(s);
    }
    const int task1 = tid + TPB;
    {
      int m = task1 / D_DIM, k = task1 % D_DIM;   // task 1: < 640 < 800
      float s = 0.f;
#pragma unroll
      for (int j = 0; j < D_DIM; ++j) s += Gt[m][k * D_DIM + j] * hcur[m][j];
      r1 = tanhf(s);
    }
    const int task2 = tid + 2 * TPB;
    if (task2 < MT * D_DIM) {
      int m = task2 / D_DIM, k = task2 % D_DIM;
      float s = 0.f;
#pragma unroll
      for (int j = 0; j < D_DIM; ++j) s += Gt[m][k * D_DIM + j] * hcur[m][j];
      r2 = tanhf(s);
    }
    __syncthreads();   // all reads of hcur done
    hcur[tid / D_DIM][tid % D_DIM] = r0;
    hcur[task1 / D_DIM][task1 % D_DIM] = r1;
    if (task2 < MT * D_DIM) hcur[task2 / D_DIM][task2 % D_DIM] = r2;
    // next iteration's barriers order these writes before the next matvec reads
  }
  __syncthreads();

  // ---- output layer: logits + sigmoid ----
  if (tid < MT * C_DIM) {
    int m = tid / C_DIM, c = tid % C_DIM;
    float s = bout[c];
#pragma unroll
    for (int k = 0; k < D_DIM; ++k) s += Wout[c * D_DIM + k] * hcur[m][k];
    out[(b0 + m) * C_DIM + c] = 1.f / (1.f + expf(-s));
  }
}

// ---------------------------------------------------------------------------
extern "C" void kernel_launch(void* const* d_in, const int* in_sizes, int n_in,
                              void* d_out, int out_size, void* d_ws, size_t ws_size,
                              hipStream_t stream) {
  const int*   words = (const int*)d_in[0];
  // d_in[1] = batch_size scalar (4096), fixed at compile time here
  const float* emb   = (const float*)d_in[2];
  const float* A     = (const float*)d_in[3];
  const float* Wout  = (const float*)d_in[4];
  const float* bout  = (const float*)d_in[5];
  float*       out   = (float*)d_out;

  _Float16* Abf = (_Float16*)d_ws;   // 10*40*32*16 halfs = 409,600 bytes

  const int total = KCH * NTILES * 32 * 16;
  repack_A<<<(total + 255) / 256, 256, 0, stream>>>(A, Abf);
  rnn_fused<<<B_BATCH / MT, TPB, 0, stream>>>(words, emb, Abf, Wout, bout, out);
}
